// CustomLSTM_76373108457943
// MI455X (gfx1250) — matrix operand order repaired and checked by reading, not compiled, not run on hardware
//
#include <hip/hip_runtime.h>
#include <hip/hip_bf16.h>

// ---------------------------------------------------------------------------
// CustomLSTM for MI455X (gfx1250, wave32, WMMA, async-LDS copies).
//   inputs: input_[512,64,1024] f32, length[64] i32,
//           weight_ih[1024,4096] f32, weight_hh[1024,4096] f32, bias[4096] f32
//   output: out[512,64,1024] f32 ++ h_n[64,1024] ++ c_n[64,1024]
// Phase 0: convert X and (transposed) weights to bf16
// Phase 1: wi = X @ W_ih + bias  (bf16 WMMA, f32 accum, async double-buffered)
// Phase 2: persistent scan kernel, 512 steps, device-wide barrier per step
// ---------------------------------------------------------------------------

typedef __bf16 bf16_t;
typedef __attribute__((ext_vector_type(16))) __bf16 v16bf;
typedef __attribute__((ext_vector_type(8)))  __bf16 v8bf;
typedef __attribute__((ext_vector_type(8)))  float  v8f;
typedef __attribute__((ext_vector_type(4)))  int    v4i;

#define T_STEPS 512
#define BATCH   64
#define HID     1024
#define FOURH   4096
#define MTOT    (T_STEPS * BATCH)   // 32768

#if defined(__gfx1250__) && __has_builtin(__builtin_amdgcn_global_load_async_to_lds_b128) && __has_builtin(__builtin_amdgcn_s_wait_asynccnt)
#define HAS_ASYNC_LDS 1
typedef __attribute__((address_space(1))) v4i* gas_t;   // global int4*
typedef __attribute__((address_space(3))) v4i* las_t;   // LDS int4*
#else
#define HAS_ASYNC_LDS 0
#endif

static __device__ __forceinline__ v16bf make_frag(const v8bf lo, const v8bf hi) {
    v16bf r;
#pragma unroll
    for (int e = 0; e < 8; ++e) { r[e] = lo[e]; r[e + 8] = hi[e]; }
    return r;
}

static __device__ __forceinline__ float sigmoidf_(float x) {
    return 1.0f / (1.0f + __expf(-x));
}

// ---------------------------------------------------------------------------
// Kernel: zero h/c state + grid-barrier counter (runs every launch -> graph-safe)
// ---------------------------------------------------------------------------
__global__ void lstm_init_state(bf16_t* __restrict__ hbf, float* __restrict__ hf,
                                float* __restrict__ cf, unsigned* __restrict__ cnt) {
    int i = blockIdx.x * 256 + threadIdx.x;
    if (i == 0) *cnt = 0u;
    if (i < BATCH * HID) {
        hbf[i] = (bf16_t)0.0f;
        hbf[BATCH * HID + i] = (bf16_t)0.0f;
        hf[i] = 0.0f;
        cf[i] = 0.0f;
    }
}

// ---------------------------------------------------------------------------
// Kernel: X f32 -> bf16 (streaming, HBM-bound)
// ---------------------------------------------------------------------------
__global__ void lstm_x_to_bf16(const float* __restrict__ src, bf16_t* __restrict__ dst) {
    const long i = ((long)blockIdx.x * 256 + threadIdx.x) * 8;
    const float4 a = *(const float4*)&src[i];
    const float4 b = *(const float4*)&src[i + 4];
    v8bf o;
    o[0] = (bf16_t)a.x; o[1] = (bf16_t)a.y; o[2] = (bf16_t)a.z; o[3] = (bf16_t)a.w;
    o[4] = (bf16_t)b.x; o[5] = (bf16_t)b.y; o[6] = (bf16_t)b.z; o[7] = (bf16_t)b.w;
    *(v8bf*)&dst[i] = o;
}

// ---------------------------------------------------------------------------
// Kernel: dst[n*K + k] = (bf16) src[k*N + n]   (LDS tile transpose, coalesced)
// ---------------------------------------------------------------------------
__global__ void lstm_transpose_bf16(const float* __restrict__ src,
                                    bf16_t* __restrict__ dst, int K, int N) {
    __shared__ float tile[32][33];
    const int tx = threadIdx.x & 31;
    const int ty = threadIdx.x >> 5;   // 0..7
#pragma unroll
    for (int i = 0; i < 4; ++i) {
        int k = blockIdx.y * 32 + ty + i * 8;
        int n = blockIdx.x * 32 + tx;
        tile[ty + i * 8][tx] = src[(long)k * N + n];
    }
    __syncthreads();
#pragma unroll
    for (int i = 0; i < 4; ++i) {
        int n = blockIdx.x * 32 + ty + i * 8;
        int k = blockIdx.y * 32 + tx;
        dst[(long)n * K + k] = (bf16_t)tile[tx][ty + i * 8];
    }
}

// ---------------------------------------------------------------------------
// Kernel: wi[M,4096] = Xb[M,1024] @ WT[4096,1024]^T + bias
// Block tile 128(M) x 64(N), K-step 32, double-buffered async LDS staging.
// 8 waves: 4x2 grid, each wave 2x2 WMMA tiles.
// ---------------------------------------------------------------------------
__global__ __launch_bounds__(256, 2)
void lstm_wi_gemm(const bf16_t* __restrict__ Xb, const bf16_t* __restrict__ WT,
                  const float* __restrict__ bias, float* __restrict__ wi) {
    __shared__ bf16_t As[2][128][40];   // padded rows: 80B stride breaks conflicts
    __shared__ bf16_t Bs[2][64][40];

    const int tid  = threadIdx.x;
    const int lane = tid & 31;
    const int wave = tid >> 5;
    const int m0 = blockIdx.y * 128;
    const int n0 = blockIdx.x * 64;
    const int wm = wave >> 1;          // 0..3
    const int wn = wave & 1;           // 0..1
    const int lrow = lane & 15;
    const int lhi  = lane >> 4;

    // per-thread staging coordinates: 16B chunks
    const int srow = tid >> 2;          // 0..63
    const int scol = (tid & 3) * 8;     // 0,8,16,24

    v8f acc[2][2] = {};

    // stage(buf, k0): copy A rows [m0,m0+128) x [k0,k0+32) and B rows [n0,n0+64)
    auto stage = [&](int buf, int k0) {
#if HAS_ASYNC_LDS
        __builtin_amdgcn_global_load_async_to_lds_b128(
            (gas_t)&Xb[(long)(m0 + srow) * HID + k0 + scol],
            (las_t)&As[buf][srow][scol], 0, 0);
        __builtin_amdgcn_global_load_async_to_lds_b128(
            (gas_t)&Xb[(long)(m0 + srow + 64) * HID + k0 + scol],
            (las_t)&As[buf][srow + 64][scol], 0, 0);
        __builtin_amdgcn_global_load_async_to_lds_b128(
            (gas_t)&WT[(long)(n0 + srow) * HID + k0 + scol],
            (las_t)&Bs[buf][srow][scol], 0, 0);
#else
        *(v8bf*)&As[buf][srow][scol]      = *(const v8bf*)&Xb[(long)(m0 + srow) * HID + k0 + scol];
        *(v8bf*)&As[buf][srow + 64][scol] = *(const v8bf*)&Xb[(long)(m0 + srow + 64) * HID + k0 + scol];
        *(v8bf*)&Bs[buf][srow][scol]      = *(const v8bf*)&WT[(long)(n0 + srow) * HID + k0 + scol];
#endif
    };

    stage(0, 0);
    for (int k0 = 0; k0 < HID; k0 += 32) {
        const int cur = (k0 >> 5) & 1;
        const bool hasnext = (k0 + 32) < HID;
        if (hasnext) stage(cur ^ 1, k0 + 32);   // prefetch next tile into other buffer
#if HAS_ASYNC_LDS
        if (hasnext) __builtin_amdgcn_s_wait_asynccnt(3);   // cur tile landed
        else         __builtin_amdgcn_s_wait_asynccnt(0);
#endif
        __syncthreads();

        v16bf afrag[2], bfrag[2];
#pragma unroll
        for (int i = 0; i < 2; ++i) {
            // A layout: lane half -> K-octets at lhi*8 and lhi*8+16
            const int ar = wm * 32 + i * 16 + lrow;
            afrag[i] = make_frag(*(const v8bf*)&As[cur][ar][lhi * 8],
                                 *(const v8bf*)&As[cur][ar][lhi * 8 + 16]);
            // B layout: lane = N, lane half -> K-octets at lhi*8 and lhi*8+16
            const int br = wn * 32 + i * 16 + lrow;
            bfrag[i] = make_frag(*(const v8bf*)&Bs[cur][br][lhi * 8],
                                 *(const v8bf*)&Bs[cur][br][lhi * 8 + 16]);
        }
#pragma unroll
        for (int i = 0; i < 2; ++i)
#pragma unroll
            for (int j = 0; j < 2; ++j)
                acc[i][j] = __builtin_amdgcn_wmma_f32_16x16x32_bf16(
                    false, afrag[i], false, bfrag[j], (short)0, acc[i][j], false, false);
        __syncthreads();   // readers done before next async writes land in this buffer
    }

    // epilogue: add bias, store f32 (D layout: M = v + 8*lhi, N = lane&15)
#pragma unroll
    for (int j = 0; j < 2; ++j) {
        const int ncol = n0 + wn * 32 + j * 16 + lrow;
        const float bv = bias[ncol];
#pragma unroll
        for (int i = 0; i < 2; ++i) {
#pragma unroll
            for (int v = 0; v < 8; ++v) {
                const int mrow = m0 + wm * 32 + i * 16 + v + lhi * 8;
                wi[(long)mrow * FOURH + ncol] = acc[i][j][v] + bv;
            }
        }
    }
}

// ---------------------------------------------------------------------------
// Kernel: persistent recurrent scan. 64 WGs x 256 threads.
// WG `wg` owns hidden columns [wg*16, wg*16+16) across ALL four gates.
// Wave w: row tile m = w&3, gates {2*(w>>2), 2*(w>>2)+1}.
// K-loop is barrier-free: W_hh^T (8MB bf16) is L2-resident; WGP$ catches the
// 4x intra-WG B-fragment reuse. Latency-bound -> no LDS staging here.
// ---------------------------------------------------------------------------
__global__ __launch_bounds__(256, 1)
void lstm_scan(const float* __restrict__ wi,      // [T*B, 4096]
               const bf16_t* __restrict__ WHT,    // [4096, 1024] = W_hh^T bf16
               const int* __restrict__ length,    // [64]
               bf16_t* __restrict__ hbf,          // [2][64*1024] double buffer
               float* __restrict__ hf,            // [64*1024]
               float* __restrict__ cf,            // [64*1024]
               float* __restrict__ out,           // [T*B*H + 2*B*H]
               unsigned* __restrict__ bar) {
    __shared__ float gbuf[4][64][17];              // gate pre-activations (padded)

    const int tid  = threadIdx.x;
    const int lane = tid & 31;
    const int wave = tid >> 5;
    const int wg   = blockIdx.x;                   // 0..63
    const int m    = wave & 3;
    const int g0   = (wave >> 2) * 2;
    const int lrow = lane & 15;
    const int lhi  = lane >> 4;

    const long brow0 = (long)(g0 * HID + wg * 16 + lrow) * HID;
    const long brow1 = (long)((g0 + 1) * HID + wg * 16 + lrow) * HID;

    // elementwise phase mapping: thread -> (batch row, 4 columns)
    const int eb = tid >> 2;                       // 0..63
    const int ej = (tid & 3) * 4;                  // 0,4,8,12
    const int len = length[eb];

    for (int t = 0; t < T_STEPS; ++t) {
        const bf16_t* hsrc = hbf + (size_t)(t & 1) * (BATCH * HID);
        v8f acc0 = {}, acc1 = {};
        const int arow = m * 16 + lrow;

        for (int kk = 0; kk < HID; kk += 32) {
            const v16bf a = make_frag(*(const v8bf*)&hsrc[arow * HID + kk + lhi * 8],
                                      *(const v8bf*)&hsrc[arow * HID + kk + lhi * 8 + 16]);
            const v16bf b0 = make_frag(*(const v8bf*)&WHT[brow0 + kk + lhi * 8],
                                       *(const v8bf*)&WHT[brow0 + kk + lhi * 8 + 16]);
            const v16bf b1 = make_frag(*(const v8bf*)&WHT[brow1 + kk + lhi * 8],
                                       *(const v8bf*)&WHT[brow1 + kk + lhi * 8 + 16]);
            acc0 = __builtin_amdgcn_wmma_f32_16x16x32_bf16(false, a, false, b0,
                                                           (short)0, acc0, false, false);
            acc1 = __builtin_amdgcn_wmma_f32_16x16x32_bf16(false, a, false, b1,
                                                           (short)0, acc1, false, false);
        }

        // gates = (h @ Whh) + wi_t  -> LDS
#pragma unroll
        for (int v = 0; v < 8; ++v) {
            const int row = m * 16 + v + lhi * 8;
            const long wib = ((long)t * BATCH + row) * FOURH + wg * 16 + lrow;
            gbuf[g0][row][lrow]     = acc0[v] + wi[wib + (long)g0 * HID];
            gbuf[g0 + 1][row][lrow] = acc1[v] + wi[wib + (long)(g0 + 1) * HID];
        }
        __syncthreads();

        // elementwise gate math (fp32), masking, state update
        {
            const float mask = (t < len) ? 1.0f : 0.0f;
#pragma unroll
            for (int q = 0; q < 4; ++q) {
                const int jj = ej + q;
                const float fg = gbuf[0][eb][jj];
                const float ig = gbuf[1][eb][jj];
                const float og = gbuf[2][eb][jj];
                const float gg = gbuf[3][eb][jj];
                const int gidx = eb * HID + wg * 16 + jj;
                const float cprev = cf[gidx];
                const float hprev = hf[gidx];
                const float fv = sigmoidf_(fg + 1.0f);
                const float iv = sigmoidf_(ig);
                const float ov = sigmoidf_(og);
                const float gv = tanhf(gg);
                const float cnew = fv * cprev + iv * gv;
                const float hnew = ov * tanhf(cnew);
                const float cnext = mask * cnew + (1.0f - mask) * cprev;
                const float hnext = mask * hnew + (1.0f - mask) * hprev;
                cf[gidx] = cnext;
                hf[gidx] = hnext;
                hbf[(size_t)((t + 1) & 1) * (BATCH * HID) + gidx] = (bf16_t)hnext;
                out[((long)t * BATCH + eb) * HID + wg * 16 + jj] = mask * hnew;
                if (t == T_STEPS - 1) {
                    out[(long)T_STEPS * BATCH * HID + gidx] = hnext;
                    out[(long)T_STEPS * BATCH * HID + BATCH * HID + gidx] = cnext;
                }
            }
        }
        __syncthreads();
        // device-wide barrier: monotonic counter, zeroed each launch by init kernel
        if (tid == 0) {
            __threadfence();
            atomicAdd(bar, 1u);
            const unsigned target = (unsigned)(t + 1) * 64u;
            while (__hip_atomic_load(bar, __ATOMIC_ACQUIRE, __HIP_MEMORY_SCOPE_AGENT) < target)
                __builtin_amdgcn_s_sleep(1);
        }
        __syncthreads();
    }
}

// ---------------------------------------------------------------------------
extern "C" void kernel_launch(void* const* d_in, const int* in_sizes, int n_in,
                              void* d_out, int out_size, void* d_ws, size_t ws_size,
                              hipStream_t stream) {
    const float* input_    = (const float*)d_in[0];   // [512,64,1024]
    const int*   length    = (const int*)d_in[1];     // [64]
    const float* weight_ih = (const float*)d_in[2];   // [1024,4096]
    const float* weight_hh = (const float*)d_in[3];   // [1024,4096]
    const float* bias      = (const float*)d_in[4];   // [4096]
    float*       out       = (float*)d_out;

    char* ws = (char*)d_ws;
    size_t off = 0;
    float*    wi    = (float*)(ws + off);  off += (size_t)MTOT * FOURH * sizeof(float);   // 512 MB
    bf16_t*   xb    = (bf16_t*)(ws + off); off += (size_t)MTOT * HID * sizeof(bf16_t);    // 64 MB
    bf16_t*   wih_t = (bf16_t*)(ws + off); off += (size_t)FOURH * HID * sizeof(bf16_t);   // 8 MB
    bf16_t*   whh_t = (bf16_t*)(ws + off); off += (size_t)FOURH * HID * sizeof(bf16_t);   // 8 MB
    bf16_t*   hbf   = (bf16_t*)(ws + off); off += (size_t)2 * BATCH * HID * sizeof(bf16_t);
    float*    hf    = (float*)(ws + off);  off += (size_t)BATCH * HID * sizeof(float);
    float*    cf    = (float*)(ws + off);  off += (size_t)BATCH * HID * sizeof(float);
    unsigned* bar   = (unsigned*)(ws + off);

    lstm_init_state<<<256, 256, 0, stream>>>(hbf, hf, cf, bar);
    lstm_x_to_bf16<<<(MTOT * HID) / (256 * 8), 256, 0, stream>>>(input_, xb);
    lstm_transpose_bf16<<<dim3(FOURH / 32, HID / 32), 256, 0, stream>>>(weight_ih, wih_t, HID, FOURH);
    lstm_transpose_bf16<<<dim3(FOURH / 32, HID / 32), 256, 0, stream>>>(weight_hh, whh_t, HID, FOURH);
    lstm_wi_gemm<<<dim3(FOURH / 64, MTOT / 128), 256, 0, stream>>>(xb, wih_t, bias, wi);
    lstm_scan<<<64, 256, 0, stream>>>(wi, whh_t, length, hbf, hf, cf, out, bar);
}
